// Matcher_67319317397932
// MI455X (gfx1250) — matrix operand order repaired and checked by reading, not compiled, not run on hardware
//
#include <hip/hip_runtime.h>
#include <hip/hip_bf16.h>
#include <math.h>

// ---------------------------------------------------------------------------
// DETR-style greedy matcher for MI455X (gfx1250, wave32, WMMA).
// Phase 1 (store-bandwidth-bound, ~92MB @ 23.3TB/s => ~4us budget):
//   * one block per 16-query row panel; sigmoid probs staged once in LDS
//     (bf16, fetched back as A fragments via ds_load_b128)
//   * negated one-hot B fragments precomputed ONCE into a 307KB L2-resident
//     table (identical for all 900 row panels) -> inner loop is one clause of
//     6 b128 loads + 3 back-to-back v_wmma_f32_16x16x32_bf16
//   * fused L1-bbox + GIoU epilogue with v_rcp_f32 (no IEEE div chains)
// Phase 2: per-image greedy argmin matching with an LDS row-min cache;
//   re-min reads hit L2 (whole 92MB cost matrix fits in 192MB L2).
// ---------------------------------------------------------------------------

#define NB   16                  // batch
#define NQ   900                 // queries per image
#define NC   91                  // classes
#define NCP  96                  // classes padded to 3*32 for WMMA K
#define NT   100                 // targets per image
#define NBT  (NB*NT)             // 1600 total target columns
#define NBQ  (NB*NQ)             // 14400 total query rows
#define NCT  (NBT/16)            // 100 column tiles
#define COST_ELEMS ((long)NB*NQ*NBT)               // 23,040,000
#define BTAB_BYTES ((size_t)NCT*3*32*16*2)         // 307,200 bytes

typedef __attribute__((ext_vector_type(16))) __bf16 v16bf;
typedef __attribute__((ext_vector_type(8)))  __bf16 v8bf;
typedef __attribute__((ext_vector_type(8)))  float  v8f;

__device__ __forceinline__ float fast_rcp(float x) {
  return __builtin_amdgcn_rcpf(x);          // v_rcp_f32, ~1ulp, no div chain
}
__device__ __forceinline__ float sigmoid_fast(float x) {
  return fast_rcp(1.0f + __expf(-x));
}

// --------------------------------------------------------------------------
// Precompute per-lane negated-one-hot B fragments for all 100 column tiles.
// Layout: btab[((ct*3 + s)*32 + lane) * 16 .. +15] = 16 bf16 halves (32B).
// B 32x16 layout used here (contiguous k, matches A): lanes 0-15 hold K=0..15, lanes 16-31 hold K=16..31.
// --------------------------------------------------------------------------
__global__ void __launch_bounds__(96)
matcher_btab_kernel(const int* __restrict__ tgt_labels,  // [NBT]
                    __bf16* __restrict__ btab)
{
  const int ct    = blockIdx.x;           // column tile 0..99
  const int s     = threadIdx.x >> 5;     // k-step 0..2
  const int lane  = threadIdx.x & 31;
  const int group = lane >> 4;
  const int ln    = lane & 15;
  const int tid_cls = tgt_labels[ct * 16 + ln];

  v16bf b;
  #pragma unroll
  for (int h = 0; h < 16; ++h) {
    const int k = s * 32 + group * 16 + h;
    b[h] = (tid_cls == k) ? (__bf16)(-1.0f) : (__bf16)(0.0f);
  }
  *(v16bf*)&btab[(size_t)((ct * 3 + s) * 32 + lane) * 16] = b;
}

// One block (8 waves) per 16-row panel of the [14400 x 1600] cost matrix.
__global__ void __launch_bounds__(256)
matcher_cost_kernel(const float* __restrict__ logits,      // [NBQ, 91]
                    const float* __restrict__ pred_boxes,  // [NBQ, 4] cxcywh
                    const int*   __restrict__ tgt_labels,  // [NBT]
                    const float* __restrict__ tgt_boxes,   // [NBT, 4] cxcywh
                    const __bf16* __restrict__ btab,       // table or nullptr
                    float*       __restrict__ cost)        // [NBQ, NBT]
{
  __shared__ alignas(16) __bf16 sprob[16 * NCP];   // 16 rows x 96 classes, 3KB

  const int tid   = threadIdx.x;
  const int lane  = tid & 31;
  const int wave  = tid >> 5;
  const int group = lane >> 4;                 // half-wave group
  const int ln    = lane & 15;
  const int i0    = blockIdx.x * 16;           // query-row base of this panel

  // ---- stage sigmoid(logits) for the whole panel into LDS (once) ----------
  #pragma unroll
  for (int u = 0; u < (16 * NCP) / 256; ++u) {
    const int idx = tid + u * 256;
    const int m = idx / NCP, k = idx % NCP;
    float v = 0.0f;
    if (k < NC) v = sigmoid_fast(logits[(long)(i0 + m) * NC + k]);
    sprob[idx] = (__bf16)v;
  }
  // Pull the next panel's logits toward the caches while we compute.
  if (blockIdx.x + 1 < NBQ / 16)
    __builtin_prefetch(logits + (long)(i0 + 16) * NC + tid, 0, 1);
  __syncthreads();

  // ---- hoist A fragments (row-panel invariant) from LDS --------------------
  // A 16x32 layout used here (contiguous k, as in btab): lane m=ln; group 0 holds K=0..15(+32s),
  // group 1 holds K=16..31(+32s) -> two aligned 16B LDS loads each.
  v16bf afrag[3];
  #pragma unroll
  for (int s = 0; s < 3; ++s) {
    const v8bf lo = *(const v8bf*)&sprob[ln * NCP + s * 32 + group * 16];
    const v8bf hi = *(const v8bf*)&sprob[ln * NCP + s * 32 + group * 16 + 8];
    #pragma unroll
    for (int h = 0; h < 8; ++h) { afrag[s][h] = lo[h]; afrag[s][8 + h] = hi[h]; }
  }

  // ---- hoist pred boxes for this lane's 8 rows (M = r + 8*group) -----------
  float pcx[8], pcy[8], pw[8], ph[8];
  float px1[8], py1[8], px2[8], py2[8], areaP[8];
  #pragma unroll
  for (int r = 0; r < 8; ++r) {
    const int i = i0 + r + 8 * group;
    const float4 pb = ((const float4*)pred_boxes)[i];
    pcx[r] = pb.x; pcy[r] = pb.y; pw[r] = pb.z; ph[r] = pb.w;
    px1[r] = pb.x - 0.5f * pb.z; py1[r] = pb.y - 0.5f * pb.w;
    px2[r] = pb.x + 0.5f * pb.z; py2[r] = pb.y + 0.5f * pb.w;
    areaP[r] = pb.z * pb.w;
  }

  // ---- epilogue shared by both paths ---------------------------------------
  auto epilogue = [&](int ctile, const v8f& acc) {
    const int j = ctile * 16 + ln;
    const float4 tb = ((const float4*)tgt_boxes)[j];
    const float tx1 = tb.x - 0.5f * tb.z, ty1 = tb.y - 0.5f * tb.w;
    const float tx2 = tb.x + 0.5f * tb.z, ty2 = tb.y + 0.5f * tb.w;
    const float areaT = tb.z * tb.w;
    #pragma unroll
    for (int r = 0; r < 8; ++r) {
      const int i = i0 + r + 8 * group;
      const float l1 = fabsf(pcx[r] - tb.x) + fabsf(pcy[r] - tb.y) +
                       fabsf(pw[r]  - tb.z) + fabsf(ph[r]  - tb.w);
      const float ix1 = fmaxf(px1[r], tx1), iy1 = fmaxf(py1[r], ty1);
      const float ix2 = fminf(px2[r], tx2), iy2 = fminf(py2[r], ty2);
      const float inter = fmaxf(ix2 - ix1, 0.0f) * fmaxf(iy2 - iy1, 0.0f);
      const float uni   = areaP[r] + areaT - inter;
      const float iou   = inter * fast_rcp(uni);
      const float ex1 = fminf(px1[r], tx1), ey1 = fminf(py1[r], ty1);
      const float ex2 = fmaxf(px2[r], tx2), ey2 = fmaxf(py2[r], ty2);
      const float areaE = (ex2 - ex1) * (ey2 - ey1);   // >= 0 by construction
      const float giou  = iou - (areaE - uni) * fast_rcp(areaE);
      // cost = l1 + (-prob) + (-giou); acc already holds -prob.
      cost[(long)i * NBT + j] = l1 + acc[r] - giou;
    }
  };

  // ---- sweep column tiles: wave w handles tiles w, w+8, ... ----------------
  if (btab != nullptr) {
    // Fast path: all 6 B-fragment loads issued up front (one clause, one
    // s_wait_loadcnt), then 3 back-to-back WMMAs on the matrix pipe.
    for (int ct = wave; ct < NCT; ct += 8) {
      v16bf b[3];
      #pragma unroll
      for (int s = 0; s < 3; ++s)
        b[s] = *(const v16bf*)&btab[(size_t)((ct * 3 + s) * 32 + lane) * 16];
      v8f acc = {};
      #pragma unroll
      for (int s = 0; s < 3; ++s)
        acc = __builtin_amdgcn_wmma_f32_16x16x32_bf16(
            false, afrag[s], false, b[s], (short)0, acc, false, false);
      epilogue(ct, acc);
    }
  } else {
    // Fallback: build the one-hot in VALU (no workspace available).
    for (int ct = wave; ct < NCT; ct += 8) {
      const int tid_cls = tgt_labels[ct * 16 + ln];
      v8f acc = {};
      #pragma unroll
      for (int s = 0; s < 3; ++s) {
        v16bf b;
        #pragma unroll
        for (int h = 0; h < 16; ++h) {
          const int k = s * 32 + group * 16 + h;
          b[h] = (tid_cls == k) ? (__bf16)(-1.0f) : (__bf16)(0.0f);
        }
        acc = __builtin_amdgcn_wmma_f32_16x16x32_bf16(
            false, afrag[s], false, b, (short)0, acc, false, false);
      }
      epilogue(ct, acc);
    }
  }
}

// One workgroup per image. Sequential greedy matching with an LDS row-min
// cache; re-mins read cost from L2.
__global__ void __launch_bounds__(256)
matcher_greedy_kernel(const float* __restrict__ cost,   // [NBQ, NBT]
                      float* __restrict__ rows_out,     // [NB, NT]
                      float* __restrict__ cols_out)     // [NB, NT]
{
  __shared__ float rowmin[NQ];
  __shared__ int   rowarg[NQ];
  __shared__ int   colBusy[NT];
  __shared__ float redv[256];
  __shared__ int   redq[256];
  __shared__ int   s_bc;

  const int b   = blockIdx.x;
  const int tid = threadIdx.x;
  const float* C = cost + (long)b * NQ * NBT;
  const float INF = __builtin_inff();

  for (int c = tid; c < NT; c += 256) colBusy[c] = 0;
  // Per-row min over the first NT columns (the only valid ones).
  for (int q = tid; q < NQ; q += 256) {
    float m = INF; int arg = 0;
    for (int c = 0; c < NT; ++c) {
      const float v = C[(long)q * NBT + c];
      if (v < m) { m = v; arg = c; }   // strict < -> lowest col on ties
    }
    rowmin[q] = m; rowarg[q] = arg;
  }
  __syncthreads();

  for (int t = 0; t < NT; ++t) {
    // Parallel argmin over rowmin[0..899], lowest row on ties.
    float m = INF; int arg = NQ;
    for (int q = tid; q < NQ; q += 256) {
      const float v = rowmin[q];
      if (v < m) { m = v; arg = q; }
    }
    redv[tid] = m; redq[tid] = arg;
    __syncthreads();
    if (tid == 0) {
      float bm = INF; int bq = NQ;
      for (int u = 0; u < 256; ++u) {
        if (redv[u] < bm || (redv[u] == bm && redq[u] < bq)) {
          bm = redv[u]; bq = redq[u];
        }
      }
      const int bc = rowarg[bq];
      s_bc = bc;
      rows_out[b * NT + t] = (float)bq;
      cols_out[b * NT + t] = (float)bc;
      rowmin[bq] = INF;       // retire row
      colBusy[bc] = 1;        // retire column
    }
    __syncthreads();
    // Rows whose cached argmin was the retired column: recompute their min.
    const int bc = s_bc;
    for (int q = tid; q < NQ; q += 256) {
      if (rowmin[q] < INF && rowarg[q] == bc) {
        float m2 = INF; int a2 = 0;
        for (int c = 0; c < NT; ++c) {
          if (!colBusy[c]) {
            const float v = C[(long)q * NBT + c];
            if (v < m2) { m2 = v; a2 = c; }
          }
        }
        rowmin[q] = m2; rowarg[q] = a2;
      }
    }
    __syncthreads();
  }
}

extern "C" void kernel_launch(void* const* d_in, const int* in_sizes, int n_in,
                              void* d_out, int out_size, void* d_ws, size_t ws_size,
                              hipStream_t stream) {
  (void)in_sizes; (void)n_in; (void)out_size;
  const float* logits     = (const float*)d_in[0];   // [16,900,91]
  const float* pred_boxes = (const float*)d_in[1];   // [16,900,4]
  const int*   tgt_labels = (const int*)  d_in[2];   // [16,100]
  const float* tgt_boxes  = (const float*)d_in[3];   // [16,100,4]

  float* out      = (float*)d_out;
  float* cost     = out;                     // 23,040,000 f32
  float* rows_out = out + COST_ELEMS;        // 1,600 (as float)
  float* cols_out = rows_out + NBT;          // 1,600 (as float)

  // B-fragment table in workspace if it fits (deterministic: ws_size fixed).
  __bf16* btab = nullptr;
  if (ws_size >= BTAB_BYTES) {
    btab = (__bf16*)d_ws;
    matcher_btab_kernel<<<NCT, 96, 0, stream>>>(tgt_labels, btab);
  }

  // 900 row-panel blocks; 8 waves per block sweep the 100 column tiles.
  matcher_cost_kernel<<<NBQ / 16, 256, 0, stream>>>(
      logits, pred_boxes, tgt_labels, tgt_boxes, btab, cost);
  matcher_greedy_kernel<<<NB, 256, 0, stream>>>(cost, rows_out, cols_out);
}
